// TriangleMultiplicationIncoming_20418274525424
// MI455X (gfx1250) — matrix operand order repaired and checked by reading, not compiled, not run on hardware
//
#include <hip/hip_runtime.h>
#include <hip/hip_bf16.h>

typedef unsigned short u16;
typedef unsigned int   u32;

#define N   384
#define CC  128
#define NT  6            // N / 64
static const size_t NN = (size_t)N * (size_t)N;

// POD 128-bit chunk (safe in unions, emits b128 loads/stores)
struct alignas(16) U128 { u32 x, y, z, w; };
struct alignas(16) F4   { float x, y, z, w; };

typedef __attribute__((ext_vector_type(16))) __bf16 bf16x16;
typedef __attribute__((ext_vector_type(8)))  float  f32x8;

union Frag { bf16x16 v; U128 q[2]; };

__device__ __forceinline__ float sigmoidf(float x) {
    return 1.0f / (1.0f + __expf(-x));
}

// fp32 -> bf16 round-to-nearest-even
__device__ __forceinline__ u16 f2bf(float f) {
    u32 u = __float_as_uint(f);
    u32 r = (u + 0x7FFFu + ((u >> 16) & 1u)) >> 16;
    return (u16)r;
}

// ---------------------------------------------------------------------------
// Stage 1: layernorm(z) -> 5 linears -> a,b (gated+masked, bf16, transposed
// to [c][s][k]) + gate_o (fp32, natural layout).
// Block: 256 threads, handles 64 k-rows at one column s.
// ---------------------------------------------------------------------------
#define ZP  132   // z LDS pitch (floats), 16B-aligned rows
#define ABP 66    // a/b LDS pitch (ushorts), odd-word stride -> no bank conflicts

__global__ __launch_bounds__(256) void stage1_proj(
    const float* __restrict__ z, const float* __restrict__ mask,
    const float* __restrict__ lnw, const float* __restrict__ lnb,
    const float* __restrict__ w_pa, const float* __restrict__ b_pa,
    const float* __restrict__ w_ga, const float* __restrict__ b_ga,
    const float* __restrict__ w_pb, const float* __restrict__ b_pb,
    const float* __restrict__ w_gb, const float* __restrict__ b_gb,
    const float* __restrict__ w_go, const float* __restrict__ b_go,
    u16* __restrict__ aT, u16* __restrict__ bT, float* __restrict__ gate)
{
    __shared__ float zs[64 * ZP];
    __shared__ u16   as[128 * ABP];
    __shared__ u16   bs[128 * ABP];

    const int t  = threadIdx.x;
    const int s  = blockIdx.x % N;          // second einsum index (j for a, i for b)
    const int k0 = (blockIdx.x / N) * 64;   // k-row block

    // Load 64 z rows (coalesced: 512B per row along c)
    for (int idx = t; idx < 64 * 128; idx += 256) {
        int r = idx >> 7, c = idx & 127;
        zs[r * ZP + c] = z[((size_t)(k0 + r) * N + s) * CC + c];
    }
    __syncthreads();

    // Layernorm per row (64 threads, one row each)
    if (t < 64) {
        float m = 0.f;
        for (int c = 0; c < 128; ++c) m += zs[t * ZP + c];
        m *= (1.f / 128.f);
        float v = 0.f;
        for (int c = 0; c < 128; ++c) { float d = zs[t * ZP + c] - m; v += d * d; }
        float rs = rsqrtf(v * (1.f / 128.f) + 1e-5f);
        for (int c = 0; c < 128; ++c)
            zs[t * ZP + c] = (zs[t * ZP + c] - m) * rs * lnw[c] + lnb[c];
    }
    __syncthreads();

    const int cout = t & 127;
    const int half = t >> 7;

    for (int ch = 0; ch < 4; ++ch) {
        const int r0 = half * 32 + ch * 8;
        float apa[8] = {}, aga[8] = {}, apb[8] = {}, agb[8] = {}, ago[8] = {};

        for (int k = 0; k < 128; k += 4) {
            const F4 wpa = *(const F4*)(w_pa + cout * 128 + k);
            const F4 wga = *(const F4*)(w_ga + cout * 128 + k);
            const F4 wpb = *(const F4*)(w_pb + cout * 128 + k);
            const F4 wgb = *(const F4*)(w_gb + cout * 128 + k);
            const F4 wgo = *(const F4*)(w_go + cout * 128 + k);
#pragma unroll
            for (int rr = 0; rr < 8; ++rr) {
                const F4 x = *(const F4*)(&zs[(r0 + rr) * ZP + k]);  // LDS broadcast
                apa[rr] = fmaf(wpa.x, x.x, fmaf(wpa.y, x.y, fmaf(wpa.z, x.z, fmaf(wpa.w, x.w, apa[rr]))));
                aga[rr] = fmaf(wga.x, x.x, fmaf(wga.y, x.y, fmaf(wga.z, x.z, fmaf(wga.w, x.w, aga[rr]))));
                apb[rr] = fmaf(wpb.x, x.x, fmaf(wpb.y, x.y, fmaf(wpb.z, x.z, fmaf(wpb.w, x.w, apb[rr]))));
                agb[rr] = fmaf(wgb.x, x.x, fmaf(wgb.y, x.y, fmaf(wgb.z, x.z, fmaf(wgb.w, x.w, agb[rr]))));
                ago[rr] = fmaf(wgo.x, x.x, fmaf(wgo.y, x.y, fmaf(wgo.z, x.z, fmaf(wgo.w, x.w, ago[rr]))));
            }
        }
#pragma unroll
        for (int rr = 0; rr < 8; ++rr) {
            const int r = r0 + rr;
            const float mk = mask[(size_t)(k0 + r) * N + s];
            const float av = (apa[rr] + b_pa[cout]) * sigmoidf(aga[rr] + b_ga[cout]) * mk;
            const float bv = (apb[rr] + b_pb[cout]) * sigmoidf(agb[rr] + b_gb[cout]) * mk;
            gate[((size_t)(k0 + r) * N + s) * CC + cout] = sigmoidf(ago[rr] + b_go[cout]);
            as[cout * ABP + r] = f2bf(av);
            bs[cout * ABP + r] = f2bf(bv);
        }
    }
    __syncthreads();

    // Transposed store: aT[c][s][k], bT[c][s][k] — 64B contiguous per thread
    {
        const int cc = t >> 1, h = t & 1;
        const size_t base = (size_t)cc * NN + (size_t)s * N + k0 + h * 32;
        const u32* sa = (const u32*)&as[cc * ABP + h * 32];  // 4B-aligned
        const u32* sb = (const u32*)&bs[cc * ABP + h * 32];
#pragma unroll
        for (int q = 0; q < 4; ++q) {
            U128 ua = { sa[q*4+0], sa[q*4+1], sa[q*4+2], sa[q*4+3] };
            *(U128*)(aT + base + q * 8) = ua;
        }
#pragma unroll
        for (int q = 0; q < 4; ++q) {
            U128 ub = { sb[q*4+0], sb[q*4+1], sb[q*4+2], sb[q*4+3] };
            *(U128*)(bT + base + q * 8) = ub;
        }
    }
}

// ---------------------------------------------------------------------------
// Stage 2: ct[c][i][j] = sum_k b[k,i,c] * a[k,j,c]
//   A[m=i,k] = bT[c][i][k] (row-major)   B[k,n=j] = aT[c][j][k]
// 4 waves per block in a 2x2 macro-tile (128x128 per block): waves sharing an
// i-slab (or j-slab) hit the same operand cachelines in WGP L0 -> ~2x less L2
// traffic. Each wave: 64x64 tile, 16 f32x8 accumulators, double-buffered
// fragment loads, K-loop step 32.
// ---------------------------------------------------------------------------
__device__ __forceinline__ void load_frags(
    const u16* __restrict__ Ab, const u16* __restrict__ Bb,
    int i0, int j0, int m, int hi, int k0, Frag A[4], Frag B[4])
{
#pragma unroll
    for (int it = 0; it < 4; ++it) {
        // A 16x32 bf16: lane<16 -> K{0..7,16..23}; lane>=16 -> K{8..15,24..31}
        const u16* p = Ab + (size_t)(i0 + it * 16 + m) * N + k0 + hi * 8;
        A[it].q[0] = *(const U128*)(p);
        A[it].q[1] = *(const U128*)(p + 16);
    }
#pragma unroll
    for (int jt = 0; jt < 4; ++jt) {
        // B 32x16 bf16: lane<16 -> col m, K{0..7,16..23}; lane>=16 -> col m, K{8..15,24..31}
        const u16* p = Bb + (size_t)(j0 + jt * 16 + m) * N + k0 + hi * 8;
        B[jt].q[0] = *(const U128*)(p);
        B[jt].q[1] = *(const U128*)(p + 16);
    }
}

__device__ __forceinline__ void mma_group(f32x8 acc[4][4], Frag A[4], Frag B[4])
{
#pragma unroll
    for (int it = 0; it < 4; ++it)
#pragma unroll
        for (int jt = 0; jt < 4; ++jt)
            acc[it][jt] = __builtin_amdgcn_wmma_f32_16x16x32_bf16(
                false, A[it].v, false, B[jt].v,
                (short)0, acc[it][jt], false, false);
}

__global__ __launch_bounds__(128) void stage2_einsum_wmma(
    const u16* __restrict__ aT, const u16* __restrict__ bT,
    float* __restrict__ ct)
{
    const int lane = threadIdx.x & 31;
    const int w    = threadIdx.x >> 5;       // wave id 0..3 -> 2x2 macro-tile
    const int m  = lane & 15;
    const int hi = lane >> 4;
    const int c    = blockIdx.y;
    const int tile = blockIdx.x;             // 0..8 (3x3 grid of 128x128 blocks)
    const int i0 = (tile / 3) * 128 + (w >> 1) * 64;
    const int j0 = (tile % 3) * 128 + (w & 1) * 64;

    const u16* __restrict__ Ab = bT + (size_t)c * NN;  // [i][k]
    const u16* __restrict__ Bb = aT + (size_t)c * NN;  // [j][k]

    f32x8 acc[4][4] = {};
    Frag A0[4], B0[4], A1[4], B1[4];

    load_frags(Ab, Bb, i0, j0, m, hi, 0, A0, B0);
#pragma unroll
    for (int k0 = 0; k0 < N; k0 += 64) {
        load_frags(Ab, Bb, i0, j0, m, hi, k0 + 32, A1, B1);
        mma_group(acc, A0, B0);
        if (k0 + 64 < N)
            load_frags(Ab, Bb, i0, j0, m, hi, k0 + 64, A0, B0);
        mma_group(acc, A1, B1);
    }

    // D 16x16 f32 layout: VGPR r -> (M = r + 8*hi, N = m)
#pragma unroll
    for (int it = 0; it < 4; ++it)
#pragma unroll
        for (int jt = 0; jt < 4; ++jt)
#pragma unroll
            for (int r = 0; r < 8; ++r) {
                const int row = i0 + it * 16 + r + hi * 8;
                const int col = j0 + jt * 16 + m;
                ct[(size_t)c * NN + (size_t)row * N + col] = acc[it][jt][r];
            }
}

// ---------------------------------------------------------------------------
// Stage 3: out[i,j,:] = (LN(ct[:, i, j]) @ W_o^T + b_o) * gate[i,j,:]
// Block: 256 threads, one i-row x 64 j-columns. Channel gather via LDS.
// ---------------------------------------------------------------------------
#define CTP 65

__global__ __launch_bounds__(256) void stage3_out(
    const float* __restrict__ ct, const float* __restrict__ gate,
    const float* __restrict__ mask,
    const float* __restrict__ lnw, const float* __restrict__ lnb,
    const float* __restrict__ w_po, const float* __restrict__ b_po,
    float* __restrict__ out)
{
    __shared__ float cs[128 * CTP];  // [c][j]
    __shared__ float ns[64 * ZP];    // [j][k] normalized, k-contiguous

    const int t  = threadIdx.x;
    const int i  = blockIdx.x / NT;
    const int j0 = (blockIdx.x % NT) * 64;

    for (int idx = t; idx < 128 * 64; idx += 256) {
        int c = idx >> 6, jc = idx & 63;
        cs[c * CTP + jc] = ct[(size_t)c * NN + (size_t)i * N + j0 + jc]
                         * mask[(size_t)i * N + j0 + jc];
    }
    __syncthreads();

    if (t < 64) {  // layernorm over channels, one column each
        float m = 0.f;
        for (int c = 0; c < 128; ++c) m += cs[c * CTP + t];
        m *= (1.f / 128.f);
        float v = 0.f;
        for (int c = 0; c < 128; ++c) { float d = cs[c * CTP + t] - m; v += d * d; }
        float rs = rsqrtf(v * (1.f / 128.f) + 1e-5f);
        for (int c = 0; c < 128; ++c)
            ns[t * ZP + c] = (cs[c * CTP + t] - m) * rs * lnw[c] + lnb[c];
    }
    __syncthreads();

    const int cout = t & 127;
    const int jh   = t >> 7;

    for (int ch = 0; ch < 4; ++ch) {
        const int jb = jh * 32 + ch * 8;
        float acc[8] = {};
        for (int k = 0; k < 128; k += 4) {
            const F4 w = *(const F4*)(w_po + cout * 128 + k);
#pragma unroll
            for (int rr = 0; rr < 8; ++rr) {
                const F4 x = *(const F4*)(&ns[(jb + rr) * ZP + k]);  // broadcast
                acc[rr] = fmaf(w.x, x.x, fmaf(w.y, x.y, fmaf(w.z, x.z, fmaf(w.w, x.w, acc[rr]))));
            }
        }
        const float bias = b_po[cout];
#pragma unroll
        for (int rr = 0; rr < 8; ++rr) {
            const int j = j0 + jb + rr;
            const size_t o = ((size_t)i * N + j) * CC + cout;
            out[o] = (acc[rr] + bias) * gate[o];
        }
    }
}

// ---------------------------------------------------------------------------
extern "C" void kernel_launch(void* const* d_in, const int* in_sizes, int n_in,
                              void* d_out, int out_size, void* d_ws, size_t ws_size,
                              hipStream_t stream) {
    const float* z      = (const float*)d_in[0];
    const float* pm     = (const float*)d_in[1];
    const float* ln_in_w  = (const float*)d_in[2];
    const float* ln_in_b  = (const float*)d_in[3];
    const float* ln_out_w = (const float*)d_in[4];
    const float* ln_out_b = (const float*)d_in[5];
    const float* w_pa = (const float*)d_in[6];  const float* b_pa = (const float*)d_in[7];
    const float* w_ga = (const float*)d_in[8];  const float* b_ga = (const float*)d_in[9];
    const float* w_pb = (const float*)d_in[10]; const float* b_pb = (const float*)d_in[11];
    const float* w_gb = (const float*)d_in[12]; const float* b_gb = (const float*)d_in[13];
    const float* w_go = (const float*)d_in[14]; const float* b_go = (const float*)d_in[15];
    const float* w_po = (const float*)d_in[16]; const float* b_po = (const float*)d_in[17];
    float* out = (float*)d_out;

    // Workspace layout
    const size_t szBF = (size_t)CC * NN * sizeof(u16);    // 37,748,736 B
    const size_t szF  = (size_t)CC * NN * sizeof(float);  // 75,497,472 B
    char* ws = (char*)d_ws;
    u16*   aT   = (u16*)(ws);
    u16*   bT   = (u16*)(ws + szBF);
    float* ct   = (float*)(ws + 2 * szBF);
    float* gate = (float*)(ws + 2 * szBF + szF);

    // Stage 1: 2304 blocks x 256 threads
    stage1_proj<<<dim3((N / 64) * N), dim3(256), 0, stream>>>(
        z, pm, ln_in_w, ln_in_b,
        w_pa, b_pa, w_ga, b_ga, w_pb, b_pb, w_gb, b_gb, w_go, b_go,
        aT, bT, gate);

    // Stage 2: WMMA einsum — 2x2 waves per block, 128x128 per block per channel
    stage2_einsum_wmma<<<dim3(9, CC), dim3(128), 0, stream>>>(aT, bT, ct);

    // Stage 3: 2304 blocks x 256 threads
    stage3_out<<<dim3(N * NT), dim3(256), 0, stream>>>(
        ct, gate, pm, ln_out_w, ln_out_b, w_po, b_po, out);
}
